// FlatST_82437602279463
// MI455X (gfx1250) — matrix operand order repaired and checked by reading, not compiled, not run on hardware
//
#include <hip/hip_runtime.h>
#include <hip/hip_bf16.h>
#include <math.h>

typedef __attribute__((ext_vector_type(16))) __bf16 v16bf;
typedef __attribute__((ext_vector_type(8)))  float  v8f;

#define TPB 256

// ---------------------------------------------------------------- utilities
__global__ void fill_f32(float* __restrict__ p, float v, long n) {
  long i = (long)blockIdx.x * blockDim.x + threadIdx.x;
  if (i < n) p[i] = v;
}

__global__ void selu_ip(float* __restrict__ p, long n) {
  const float lam = 1.0507009873554805f, alp = 1.6732632423543772f;
  long i = (long)blockIdx.x * blockDim.x + threadIdx.x;
  if (i >= n) return;
  float x = p[i];
  p[i] = lam * (x > 0.f ? x : alp * (__expf(x) - 1.f));
}

__global__ void avg2(const float* __restrict__ a, const float* __restrict__ b,
                     float* __restrict__ o, long n) {
  long i = (long)blockIdx.x * blockDim.x + threadIdx.x;
  if (i < n) o[i] = 0.5f * (a[i] + b[i]);
}

// ---------------------------------------------------------------- WMMA GEMM
// C[M x Nout] = A[M x K] @ (WTRANS ? W^T : W)   (bf16 inputs, f32 accumulate)
//
// W (<=128KB) is staged ONCE per workgroup into LDS, bf16-converted and
// pre-swizzled into the exact WMMA B-operand lane layout (ISA 7.12.2), so the
// inner loop's B acquisition is two ds_load_b128 per tile: no global loads,
// no converts, no loadcnt serialization. A is software-pipelined: the next
// k-step's 4x global_load_b128 are issued before the current CT back-to-back
// v_wmma_f32_16x16x32_bf16 ops, hiding HBM latency behind the XDL pipe.
//
// LDS layout (bf16 units): idx = ((ks*ntiles + ct)*32 + lane)*16 + t,
// where b[t] for (ks,ct,lane) is B element (K = ks*32 + (t>>3)*16 + (lane>>4)*8 + (t&7),
// N = ct*16 + (lane&15)).
template<bool WTRANS, int CT>
__global__ void gemm_wmma(const float* __restrict__ A, const float* __restrict__ W,
                          float* __restrict__ C, int M, int K, int Nout,
                          int total_groups) {
  extern __shared__ __bf16 ldsW[];
  const int tid = threadIdx.x;
  const int ntiles = Nout >> 4;

  // ---- stage W -> LDS in B-operand layout (cooperative, before any return)
  for (long idx = tid; idx < (long)K * Nout; idx += TPB) {
    int t      = (int)(idx & 15);
    int lane_s = (int)((idx >> 4) & 31);
    long rest  = idx >> 9;
    int ct     = (int)(rest % ntiles);
    int ks     = (int)(rest / ntiles);
    int col    = ct * 16 + (lane_s & 15);
    int kidx   = ks * 32 + (t >> 3) * 16 + (lane_s >> 4) * 8 + (t & 7);
    float v    = WTRANS ? W[(long)col * K + kidx] : W[(long)kidx * Nout + col];
    ldsW[idx]  = (__bf16)v;
  }
  __syncthreads();

  int wave = (int)(((long)blockIdx.x * blockDim.x + tid) >> 5);
  int lane = tid & 31;
  if (wave >= total_groups) return;         // wave-uniform: EXEC stays all-ones
  int ngroups = ntiles / CT;
  int mt  = wave / ngroups;
  int nt0 = (wave % ngroups) * CT;          // first column tile
  int half = lane >> 4;
  int coll = lane & 15;
  int rrow = mt * 16 + coll;
  int row  = rrow < M ? rrow : M - 1;       // clamp tail rows for loads only
  const float* pa  = A + (long)row * K + half * 8;
  const float* pa2 = pa + 16;

  v8f acc[CT] = {};
  // software-pipelined A fetch
  float4 q0 = ((const float4*)pa)[0],  q1 = ((const float4*)pa)[1];
  float4 q2 = ((const float4*)pa2)[0], q3 = ((const float4*)pa2)[1];
  const int nks = K >> 5;
  for (int ks = 0; ks < nks; ++ks) {
    float4 p0, p1, p2, p3;
    if (ks + 1 < nks) {                     // uniform branch
      const float* na  = pa  + (ks + 1) * 32;
      const float* na2 = pa2 + (ks + 1) * 32;
      p0 = ((const float4*)na)[0];  p1 = ((const float4*)na)[1];
      p2 = ((const float4*)na2)[0]; p3 = ((const float4*)na2)[1];
    }
    v16bf a;
    a[0]=(__bf16)q0.x; a[1]=(__bf16)q0.y; a[2]=(__bf16)q0.z; a[3]=(__bf16)q0.w;
    a[4]=(__bf16)q1.x; a[5]=(__bf16)q1.y; a[6]=(__bf16)q1.z; a[7]=(__bf16)q1.w;
    a[8]=(__bf16)q2.x; a[9]=(__bf16)q2.y; a[10]=(__bf16)q2.z; a[11]=(__bf16)q2.w;
    a[12]=(__bf16)q3.x; a[13]=(__bf16)q3.y; a[14]=(__bf16)q3.z; a[15]=(__bf16)q3.w;
#pragma unroll
    for (int t2 = 0; t2 < CT; ++t2) {
      const v16bf b = *(const v16bf*)&ldsW[(((long)ks * ntiles + nt0 + t2) * 32 + lane) * 16];
      acc[t2] = __builtin_amdgcn_wmma_f32_16x16x32_bf16(false, a, false, b,
                                                        (short)0, acc[t2],
                                                        false, false);
    }
    q0 = p0; q1 = p1; q2 = p2; q3 = p3;
  }

  // ---- store: tile-row validity is wave-uniform -> uniform branch, no cmpx
  bool fullM = (mt * 16 + 16 <= M);
#pragma unroll
  for (int t2 = 0; t2 < CT; ++t2) {
    float* cp = C + (long)(mt * 16 + half * 8) * Nout + (nt0 + t2) * 16 + coll;
    if (fullM) {
#pragma unroll
      for (int r = 0; r < 8; ++r) cp[(long)r * Nout] = acc[t2][r];
    } else {
#pragma unroll
      for (int r = 0; r < 8; ++r)
        if (mt * 16 + half * 8 + r < M) cp[(long)r * Nout] = acc[t2][r];
    }
  }
}

// ---------------------------------------------------------------- row dot
// out[row] = X[row,:] . a   (one wave per row, coalesced, shuffle reduce)
__global__ void rowdot(const float* __restrict__ X, const float* __restrict__ a,
                       float* __restrict__ out, int nrows, int K) {
  int wave = (int)(((long)blockIdx.x * blockDim.x + threadIdx.x) >> 5);
  int lane = threadIdx.x & 31;
  if (wave >= nrows) return;
  float s = 0.f;
  for (int k = lane; k < K; k += 32) s += X[(long)wave * K + k] * a[k];
  for (int off = 16; off > 0; off >>= 1) s += __shfl_down(s, off, 32);
  if (lane == 0) out[wave] = s;
}

// ---------------------------------------------------------------- edge phase
__device__ __forceinline__ unsigned ford(float f) {      // order-preserving map
  unsigned b = __float_as_uint(f);
  return (b & 0x80000000u) ? ~b : (b | 0x80000000u);
}
__device__ __forceinline__ float finv(unsigned u) {
  return (u & 0x80000000u) ? __uint_as_float(u ^ 0x80000000u)
                           : __uint_as_float(~u);
}

__global__ void edge_score(const int* __restrict__ src, const int* __restrict__ dst,
                           const float* __restrict__ ss, const float* __restrict__ sd,
                           float* __restrict__ s_e, unsigned* __restrict__ mx, int E) {
  int e = blockIdx.x * blockDim.x + threadIdx.x;
  if (e >= E) return;
  float s = ss[src[e]] + sd[dst[e]];
  s = s > 0.f ? s : 0.2f * s;                       // leaky_relu(., 0.2)
  s_e[e] = s;
  atomicMax(&mx[dst[e]], ford(s));                  // global_atomic_max_u32
}

__global__ void edge_exp(const int* __restrict__ dst, float* __restrict__ s_e,
                         const unsigned* __restrict__ mx, float* __restrict__ den, int E) {
  int e = blockIdx.x * blockDim.x + threadIdx.x;
  if (e >= E) return;
  int d = dst[e];
  float ev = __expf(s_e[e] - finv(mx[d]));
  s_e[e] = ev;
  atomicAdd(&den[d], ev);
}

// out[dst,:] += (s_e/den[dst]) * xs[src,:]   (4 channels per thread)
__global__ void edge_aggr_att(const int* __restrict__ src, const int* __restrict__ dst,
                              const float* __restrict__ s_e, const float* __restrict__ den,
                              const float* __restrict__ xs, float* __restrict__ out,
                              int E, int K) {
  long idx = (long)blockIdx.x * blockDim.x + threadIdx.x;
  int per = K >> 2;
  if (idx >= (long)E * per) return;
  int e = (int)(idx / per);
  int c = (int)(idx % per) * 4;
  int sN = src[e], dN = dst[e];
  float w = s_e[e] / den[dN];
  const float4 v = *(const float4*)(xs + (long)sN * K + c);
  float* o = out + (long)dN * K + c;
  atomicAdd(o + 0, w * v.x); atomicAdd(o + 1, w * v.y);
  atomicAdd(o + 2, w * v.z); atomicAdd(o + 3, w * v.w);
}

// out[dst,:] += (1/deg_dst[dst]) * xs[src,:]
__global__ void edge_aggr_mean(const int* __restrict__ src, const int* __restrict__ dst,
                               const float* __restrict__ degd, const float* __restrict__ xs,
                               float* __restrict__ out, int E, int K) {
  long idx = (long)blockIdx.x * blockDim.x + threadIdx.x;
  int per = K >> 2;
  if (idx >= (long)E * per) return;
  int e = (int)(idx / per);
  int c = (int)(idx % per) * 4;
  int sN = src[e], dN = dst[e];
  float w = 1.f / degd[dN];
  const float4 v = *(const float4*)(xs + (long)sN * K + c);
  float* o = out + (long)dN * K + c;
  atomicAdd(o + 0, w * v.x); atomicAdd(o + 1, w * v.y);
  atomicAdd(o + 2, w * v.z); atomicAdd(o + 3, w * v.w);
}

__global__ void edge_deg(const int* __restrict__ src, const int* __restrict__ dst,
                         float* __restrict__ degs, float* __restrict__ degd, int E) {
  int e = blockIdx.x * blockDim.x + threadIdx.x;
  if (e >= E) return;
  atomicAdd(&degs[src[e]], 1.f);
  atomicAdd(&degd[dst[e]], 1.f);
}

// precompute per-edge smoothing coefficients (deg is src-based, per reference)
__global__ void edge_coeff(const int* __restrict__ src, const int* __restrict__ dst,
                           const float* __restrict__ deg,
                           float* __restrict__ n1, float* __restrict__ n2,
                           float* __restrict__ nrm,
                           const float* __restrict__ a1p, const float* __restrict__ a2p,
                           int E) {
  int e = blockIdx.x * blockDim.x + threadIdx.x;
  if (e >= E) return;
  float ds = deg[src[e]], dd = deg[dst[e]];
  float dis_s = ds > 0.f ? __powf(ds, -0.6f) : 0.f;
  float dis_d = dd > 0.f ? __powf(dd, -0.6f) : 0.f;
  n1[e] = dis_s * dis_d * a1p[0];
  n2[e] = dis_s * dis_d * a2p[0] * 2.f;
  float d2s = ds > 0.f ? rsqrtf(ds) : 0.f;
  float d2d = dd > 0.f ? rsqrtf(dd) : 0.f;
  nrm[e] = d2s * d2d * (1.f / (1.f + __logf(ds + 1.f)));
}

// out[src,:] += coef[e] * in[dst,:]
__global__ void edge_smooth(const int* __restrict__ src, const int* __restrict__ dst,
                            const float* __restrict__ coef, const float* __restrict__ in,
                            float* __restrict__ out, int E, int K) {
  long idx = (long)blockIdx.x * blockDim.x + threadIdx.x;
  int per = K >> 2;
  if (idx >= (long)E * per) return;
  int e = (int)(idx / per);
  int c = (int)(idx % per) * 4;
  int sN = src[e], dN = dst[e];
  float w = coef[e];
  const float4 v = *(const float4*)(in + (long)dN * K + c);
  float* o = out + (long)sN * K + c;
  atomicAdd(o + 0, w * v.x); atomicAdd(o + 1, w * v.y);
  atomicAdd(o + 2, w * v.z); atomicAdd(o + 3, w * v.w);
}

// ---------------------------------------------------------------- host driver
static inline int blk(long n) { return (int)((n + TPB - 1) / TPB); }

extern "C" void kernel_launch(void* const* d_in, const int* in_sizes, int n_in,
                              void* d_out, int out_size, void* d_ws, size_t ws_size,
                              hipStream_t stream) {
  const float* feats = (const float*)d_in[0];
  const int*   ei    = (const int*)d_in[1];
  const float* W1s = (const float*)d_in[2];  const float* W1d = (const float*)d_in[3];
  const float* a1s = (const float*)d_in[4];  const float* a1d = (const float*)d_in[5];
  const float* W2s = (const float*)d_in[6];  const float* W2d = (const float*)d_in[7];
  const float* a2s = (const float*)d_in[8];  const float* a2d = (const float*)d_in[9];
  const float* a3s = (const float*)d_in[10]; const float* a3d = (const float*)d_in[11];
  const float* al1 = (const float*)d_in[14]; const float* al2 = (const float*)d_in[15];

  const int HID_ = in_sizes[4];
  const int IN_  = in_sizes[2] / HID_;
  const int OUT_ = in_sizes[8];
  const int Nn   = in_sizes[0] / IN_;
  const int Ee   = in_sizes[1] / 2;
  const int* src = ei;
  const int* dst = ei + Ee;

  // ---- workspace layout (floats)
  float* ws = (float*)d_ws;
  size_t o = 0;
  const int MAXW = IN_ > HID_ ? IN_ : HID_;
  float* xs  = ws + o; o += (size_t)Nn * MAXW;   // GEMM out (reused per layer)
  float* xd  = ws + o; o += (size_t)Nn * HID_;
  float* h1  = ws + o; o += (size_t)Nn * HID_;
  float* h2  = ws + o; o += (size_t)Nn * OUT_;
  float* h3  = ws + o; o += (size_t)Nn * HID_;
  float* ss  = ws + o; o += Nn;
  float* sd  = ws + o; o += Nn;
  float* den = ws + o; o += Nn;
  float* degs= ws + o; o += Nn;
  float* degd= ws + o; o += Nn;
  unsigned* mx = (unsigned*)(ws + o); o += Nn;
  float* s_e = ws + o; o += Ee;
  float* n1  = ws + o; o += Ee;
  float* n2  = ws + o; o += Ee;
  float* nrm = ws + o; o += Ee;

  float* sm_out = (float*)d_out;                       // [N, OUT]
  float* h4     = (float*)d_out + (size_t)Nn * OUT_;   // [N, IN]

  // GEMM dispatch: CT=4 column-tiles/wave when possible (A reuse), else CT=2.
  // Dynamic LDS = K*Nout bf16 (max 64KB for 256x128).
  auto launch_gemm = [&](bool wt, const float* A, const float* W, float* C,
                         int M, int K, int Nout) {
    int ntiles = Nout / 16;
    int Mt = (M + 15) / 16;
    size_t shb = (size_t)K * Nout * sizeof(__bf16);
    if (ntiles % 4 == 0) {
      int tg = Mt * (ntiles / 4);
      if (wt) gemm_wmma<true, 4><<<blk((long)tg * 32), TPB, shb, stream>>>(A, W, C, M, K, Nout, tg);
      else    gemm_wmma<false,4><<<blk((long)tg * 32), TPB, shb, stream>>>(A, W, C, M, K, Nout, tg);
    } else {
      int tg = Mt * (ntiles / 2);
      if (wt) gemm_wmma<true, 2><<<blk((long)tg * 32), TPB, shb, stream>>>(A, W, C, M, K, Nout, tg);
      else    gemm_wmma<false,2><<<blk((long)tg * 32), TPB, shb, stream>>>(A, W, C, M, K, Nout, tg);
    }
  };

  // ================= Layer 1: GAT(feats; W1) -> selu -> h1  (IN -> HID)
  launch_gemm(false, feats, W1s, xs, Nn, IN_, HID_);
  launch_gemm(false, feats, W1d, xd, Nn, IN_, HID_);
  rowdot<<<blk((long)Nn * 32), TPB, 0, stream>>>(xs, a1s, ss, Nn, HID_);
  rowdot<<<blk((long)Nn * 32), TPB, 0, stream>>>(xd, a1d, sd, Nn, HID_);
  fill_f32<<<blk(Nn), TPB, 0, stream>>>((float*)mx, 0.f, Nn);   // ford identity
  fill_f32<<<blk(Nn), TPB, 0, stream>>>(den, 0.f, Nn);
  fill_f32<<<blk((long)Nn * HID_), TPB, 0, stream>>>(h1, 0.f, (long)Nn * HID_);
  edge_score<<<blk(Ee), TPB, 0, stream>>>(src, dst, ss, sd, s_e, mx, Ee);
  edge_exp<<<blk(Ee), TPB, 0, stream>>>(dst, s_e, mx, den, Ee);
  edge_aggr_att<<<blk((long)Ee * (HID_ / 4)), TPB, 0, stream>>>(src, dst, s_e, den, xs, h1, Ee, HID_);
  selu_ip<<<blk((long)Nn * HID_), TPB, 0, stream>>>(h1, (long)Nn * HID_);

  // ================= Layer 2: GAT(h1; W2) -> h2  (HID -> OUT)
  launch_gemm(false, h1, W2s, xs, Nn, HID_, OUT_);
  launch_gemm(false, h1, W2d, xd, Nn, HID_, OUT_);
  rowdot<<<blk((long)Nn * 32), TPB, 0, stream>>>(xs, a2s, ss, Nn, OUT_);
  rowdot<<<blk((long)Nn * 32), TPB, 0, stream>>>(xd, a2d, sd, Nn, OUT_);
  fill_f32<<<blk(Nn), TPB, 0, stream>>>((float*)mx, 0.f, Nn);
  fill_f32<<<blk(Nn), TPB, 0, stream>>>(den, 0.f, Nn);
  fill_f32<<<blk((long)Nn * OUT_), TPB, 0, stream>>>(h2, 0.f, (long)Nn * OUT_);
  edge_score<<<blk(Ee), TPB, 0, stream>>>(src, dst, ss, sd, s_e, mx, Ee);
  edge_exp<<<blk(Ee), TPB, 0, stream>>>(dst, s_e, mx, den, Ee);
  edge_aggr_att<<<blk((long)Ee * (OUT_ / 4)), TPB, 0, stream>>>(src, dst, s_e, den, xs, h2, Ee, OUT_);

  // ================= Layer 3: GAT(h2; W2^T) -> selu -> h3  (OUT -> HID)
  launch_gemm(true, h2, W2s, xs, Nn, OUT_, HID_);
  launch_gemm(true, h2, W2d, xd, Nn, OUT_, HID_);
  rowdot<<<blk((long)Nn * 32), TPB, 0, stream>>>(xs, a3s, ss, Nn, HID_);
  rowdot<<<blk((long)Nn * 32), TPB, 0, stream>>>(xd, a3d, sd, Nn, HID_);
  fill_f32<<<blk(Nn), TPB, 0, stream>>>((float*)mx, 0.f, Nn);
  fill_f32<<<blk(Nn), TPB, 0, stream>>>(den, 0.f, Nn);
  fill_f32<<<blk((long)Nn * HID_), TPB, 0, stream>>>(h3, 0.f, (long)Nn * HID_);
  edge_score<<<blk(Ee), TPB, 0, stream>>>(src, dst, ss, sd, s_e, mx, Ee);
  edge_exp<<<blk(Ee), TPB, 0, stream>>>(dst, s_e, mx, den, Ee);
  edge_aggr_att<<<blk((long)Ee * (HID_ / 4)), TPB, 0, stream>>>(src, dst, s_e, den, xs, h3, Ee, HID_);
  selu_ip<<<blk((long)Nn * HID_), TPB, 0, stream>>>(h3, (long)Nn * HID_);

  // ================= Layer 4: GAT(h3; W1^T), mean aggregation -> h4  (HID -> IN)
  launch_gemm(true, h3, W1s, xs, Nn, HID_, IN_);
  fill_f32<<<blk(Nn), TPB, 0, stream>>>(degs, 0.f, Nn);
  fill_f32<<<blk(Nn), TPB, 0, stream>>>(degd, 0.f, Nn);
  edge_deg<<<blk(Ee), TPB, 0, stream>>>(src, dst, degs, degd, Ee);
  fill_f32<<<blk((long)Nn * IN_), TPB, 0, stream>>>(h4, 0.f, (long)Nn * IN_);
  edge_aggr_mean<<<blk((long)Ee * (IN_ / 4)), TPB, 0, stream>>>(src, dst, degd, xs, h4, Ee, IN_);

  // ================= Smoothing (6 scatter rounds over OUT channels)
  edge_coeff<<<blk(Ee), TPB, 0, stream>>>(src, dst, degs, n1, n2, nrm, al1, al2, Ee);

  // xs region is now free (layer-4 aggregation done): alias 6 [N,OUT] buffers
  const size_t SB = (size_t)Nn * OUT_;
  float* b0 = xs;          float* b1 = xs + SB;     float* b2 = xs + 2 * SB;
  float* b3 = xs + 3 * SB; float* b4 = xs + 4 * SB; float* b5 = xs + 5 * SB;
  const long nsm  = (long)Nn * OUT_;
  const long esm  = (long)Ee * (OUT_ / 4);

  // s1: two iterations of scatter(n1, .) starting from h2
  fill_f32<<<blk(nsm), TPB, 0, stream>>>(b0, 0.f, nsm);
  edge_smooth<<<blk(esm), TPB, 0, stream>>>(src, dst, n1, h2, b0, Ee, OUT_);
  fill_f32<<<blk(nsm), TPB, 0, stream>>>(b1, 0.f, nsm);
  edge_smooth<<<blk(esm), TPB, 0, stream>>>(src, dst, n1, b0, b1, Ee, OUT_);
  // s2: two iterations of scatter(n2, .)
  fill_f32<<<blk(nsm), TPB, 0, stream>>>(b2, 0.f, nsm);
  edge_smooth<<<blk(esm), TPB, 0, stream>>>(src, dst, n2, h2, b2, Ee, OUT_);
  fill_f32<<<blk(nsm), TPB, 0, stream>>>(b3, 0.f, nsm);
  edge_smooth<<<blk(esm), TPB, 0, stream>>>(src, dst, n2, b2, b3, Ee, OUT_);
  // sm = 0.5*s1 + 0.5*s2
  avg2<<<blk(nsm), TPB, 0, stream>>>(b1, b3, b4, nsm);
  // dynamic smoothing: two iterations of scatter(norm, .), final into d_out
  fill_f32<<<blk(nsm), TPB, 0, stream>>>(b5, 0.f, nsm);
  edge_smooth<<<blk(esm), TPB, 0, stream>>>(src, dst, nrm, b4, b5, Ee, OUT_);
  fill_f32<<<blk(nsm), TPB, 0, stream>>>(sm_out, 0.f, nsm);
  edge_smooth<<<blk(esm), TPB, 0, stream>>>(src, dst, nrm, b5, sm_out, Ee, OUT_);
}
